// LSTMTagger_41171556499837
// MI455X (gfx1250) — compile-verified, matrix operand order checked by reading, not run on hardware
//
#include <hip/hip_runtime.h>
#include <hip/hip_bf16.h>
#include <stdint.h>

// Problem constants (match reference)
#define T_LEN   32768
#define INPDIM  770
#define XPAD    800   // 774 padded to multiple of 32
#define NGATE   256   // 2 dirs * 4*HD
#define HD      32
#define TAGS    50
#define TAGPAD  64

typedef __bf16 v16bf __attribute__((ext_vector_type(16)));
typedef float  v8f   __attribute__((ext_vector_type(8)));

__device__ __forceinline__ __bf16 f2bf(float f) {
    union { float f; uint32_t u; } v; v.f = f;
    uint32_t r = v.u + 0x7FFFu + ((v.u >> 16) & 1u);   // round to nearest even
    union { unsigned short s; __bf16 b; } o; o.s = (unsigned short)(r >> 16);
    return o.b;
}

__device__ __forceinline__ float sigm(float x) { return 1.0f / (1.0f + __expf(-x)); }

__device__ __forceinline__ v16bf ldfrag(const __bf16* p) {
    union { v16bf v; uint4 q[2]; } u;
    u.q[0] = *(const uint4*)(p);        // K run:  k32 + half*8 .. +7
    u.q[1] = *(const uint4*)(p + 16);   // K run:  k32 + 16 + half*8 .. +7
    return u.v;
}

#define WMMA_BF16(A_, B_, C_) \
    __builtin_amdgcn_wmma_f32_16x16x32_bf16(false, (A_), false, (B_), (short)0, (C_), false, false)

// ---------------------------------------------------------------------------
// 1) Build x_bf16 (T x 800): [input_vecs | dp_table[dp_in] | zero pad]
// ---------------------------------------------------------------------------
__global__ void embed_kernel(const float* __restrict__ iv,
                             const float* __restrict__ dpt,
                             const int*   __restrict__ dpin,
                             __bf16* __restrict__ xb) {
    size_t idx = (size_t)blockIdx.x * blockDim.x + threadIdx.x;
    if (idx >= (size_t)T_LEN * XPAD) return;
    int t = (int)(idx / XPAD);
    int k = (int)(idx % XPAD);
    float v = 0.0f;
    if (k < INPDIM)            v = iv[(size_t)t * INPDIM + k];
    else if (k < INPDIM + 4)   v = dpt[dpin[t] * 4 + (k - INPDIM)];
    xb[idx] = f2bf(v);
}

// ---------------------------------------------------------------------------
// 2) Generic f32 -> bf16 weight convert with K padding and row padding
// ---------------------------------------------------------------------------
__global__ void cvt_pad_kernel(const float* __restrict__ src, __bf16* __restrict__ dst,
                               int rows_total, int rows_valid, int k_src, int k_dst) {
    size_t idx = (size_t)blockIdx.x * blockDim.x + threadIdx.x;
    if (idx >= (size_t)rows_total * k_dst) return;
    int n = (int)(idx / k_dst);
    int k = (int)(idx % k_dst);
    float v = (n < rows_valid && k < k_src) ? src[(size_t)n * k_src + k] : 0.0f;
    dst[idx] = f2bf(v);
}

// ---------------------------------------------------------------------------
// 3) WMMA bf16 GEMM, 1M x 4N register blocking with software double-buffering:
//    C(MxN,f32) = A(MxK,bf16) @ B(NxK,bf16)^T + bias
//    One wave -> 16(M) x 64(N): per K-step 5 fragment loads feed 4 WMMAs.
//    Next-step fragments load into a second live buffer before the current
//    WMMAs consume the first, so global loads overlap the matrix ops.
//    ~130 VGPRs total -> no scratch spills; waves_per_eu(1) removes the
//    occupancy cap on the allocator (latency is hidden in-wave).
// ---------------------------------------------------------------------------
__global__ void __attribute__((amdgpu_waves_per_eu(1)))
gemm_bf16_wmma(const __bf16* __restrict__ A, const __bf16* __restrict__ B,
               const float* __restrict__ bias, float* __restrict__ C,
               int Mtiles /* M/16 */, int Nblk /* N/64 */, int Kiters,
               int lda, int ldb, int ldc, int nvalid) {
    int wid = (int)((blockIdx.x * blockDim.x + threadIdx.x) >> 5);
    if (wid >= Mtiles * Nblk) return;             // uniform per wave
    int lane  = threadIdx.x & 31;
    int half  = lane >> 4;        // K-half selector
    int idx16 = lane & 15;        // M row (A) / N col (B)
    int m0 = (wid / Nblk) << 4;   // 16-row M tile
    int n0 = (wid % Nblk) << 6;   // 64-wide N block

    const __bf16* pa  = A + (size_t)(m0 +       idx16) * lda + half * 8;
    const __bf16* pb0 = B + (size_t)(n0 +  0 + idx16) * ldb + half * 8;
    const __bf16* pb1 = B + (size_t)(n0 + 16 + idx16) * ldb + half * 8;
    const __bf16* pb2 = B + (size_t)(n0 + 32 + idx16) * ldb + half * 8;
    const __bf16* pb3 = B + (size_t)(n0 + 48 + idx16) * ldb + half * 8;

    v8f z = {0.f,0.f,0.f,0.f,0.f,0.f,0.f,0.f};
    v8f acc0 = z, acc1 = z, acc2 = z, acc3 = z;

    // prologue: load K-step 0
    v16bf ac  = ldfrag(pa);
    v16bf b0c = ldfrag(pb0), b1c = ldfrag(pb1), b2c = ldfrag(pb2), b3c = ldfrag(pb3);

    for (int k = 0; k < Kiters - 1; ++k) {
        pa += 32; pb0 += 32; pb1 += 32; pb2 += 32; pb3 += 32;
        // prefetch K-step k+1 into a second live buffer
        v16bf an  = ldfrag(pa);
        v16bf b0n = ldfrag(pb0), b1n = ldfrag(pb1), b2n = ldfrag(pb2), b3n = ldfrag(pb3);
        // 4 WMMAs on the current buffer while the prefetch is in flight
        acc0 = WMMA_BF16(ac, b0c, acc0);
        acc1 = WMMA_BF16(ac, b1c, acc1);
        acc2 = WMMA_BF16(ac, b2c, acc2);
        acc3 = WMMA_BF16(ac, b3c, acc3);
        ac = an; b0c = b0n; b1c = b1n; b2c = b2n; b3c = b3n;
    }
    // epilogue K-step
    acc0 = WMMA_BF16(ac, b0c, acc0);
    acc1 = WMMA_BF16(ac, b1c, acc1);
    acc2 = WMMA_BF16(ac, b2c, acc2);
    acc3 = WMMA_BF16(ac, b3c, acc3);

    v8f* accs[4] = { &acc0, &acc1, &acc2, &acc3 };
#pragma unroll
    for (int nt = 0; nt < 4; ++nt) {
        int n = n0 + nt * 16 + idx16;
        float bv = (n < nvalid) ? bias[n] : 0.0f;
        v8f a = *accs[nt];
#pragma unroll
        for (int r = 0; r < 8; ++r) {
            int m = m0 + r + half * 8;
            C[(size_t)m * ldc + n] = (n < nvalid) ? (a[r] + bv) : 0.0f;
        }
    }
}

// ---------------------------------------------------------------------------
// 4) Sequential LSTM scan: ONE WAVE per direction (barriers become S_NOP on
//    single-wave workgroups). Lane j owns gate rows {j, 32+j, 64+j, 96+j},
//    so the c/h update is entirely lane-local; the only cross-lane exchange
//    is h broadcast through 32 floats of LDS (b128 broadcast reads).
//    Recurrent dot runs in separate accumulators so the global G loads'
//    latency hides under the 32-deep FMA chains.
// ---------------------------------------------------------------------------
__global__ void lstm_scan_kernel(const float* __restrict__ G,
                                 const float* __restrict__ Whh,
                                 __bf16* __restrict__ Hb, int Tlen) {
    __shared__ float h_s[HD];
    const int dir = blockIdx.x;
    const int j   = threadIdx.x;                 // 0..31
    const float* Wd = Whh + (size_t)dir * 128 * HD;

    float wi[HD], wf[HD], wg[HD], wo[HD];
#pragma unroll
    for (int k = 0; k < HD; ++k) {
        wi[k] = Wd[(size_t)(       j) * HD + k];
        wf[k] = Wd[(size_t)(HD   + j) * HD + k];
        wg[k] = Wd[(size_t)(2*HD + j) * HD + k];
        wo[k] = Wd[(size_t)(3*HD + j) * HD + k];
    }
    h_s[j] = 0.0f;
    float c = 0.0f;
    __syncthreads();

    for (int s = 0; s < Tlen; ++s) {
        int t = dir ? (Tlen - 1 - s) : s;
        const float* gp = G + (size_t)t * NGATE + dir * 128;
        float gi = gp[j], gf = gp[HD + j], gg = gp[2*HD + j], go = gp[3*HD + j];
        float di = 0.f, df = 0.f, dg = 0.f, dq = 0.f;
#pragma unroll
        for (int k = 0; k < HD; ++k) {           // 4 independent FMA chains
            float hk = h_s[k];
            di += wi[k] * hk;
            df += wf[k] * hk;
            dg += wg[k] * hk;
            dq += wo[k] * hk;
        }
        gi += di; gf += df; gg += dg; go += dq;
        c = sigm(gf) * c + sigm(gi) * tanhf(gg);
        float h = sigm(go) * tanhf(c);
        Hb[(size_t)t * (2 * HD) + dir * HD + j] = f2bf(h);
        __syncthreads();                         // S_NOP + waitcnt (1 wave)
        h_s[j] = h;
        __syncthreads();
    }
}

// ---------------------------------------------------------------------------
// 5) mask/lambda fixup -> relu -> softmax.  One thread per row.
// ---------------------------------------------------------------------------
__global__ void tag_softmax_kernel(const float* __restrict__ tag,
                                   const int* __restrict__ mask,
                                   float* __restrict__ out) {
    int t = blockIdx.x * blockDim.x + threadIdx.x;
    if (t >= T_LEN) return;
    float v[TAGS];
    const float* row = tag + (size_t)t * TAGPAD;
#pragma unroll
    for (int i = 0; i < TAGS; ++i) v[i] = row[i];
    v[TAGS - 1] = (mask[t] > 0) ? fminf(v[TAGS - 1], 1.0f) : 1.0f;  // LAMBDA=1.0
    float mx = 0.0f;
#pragma unroll
    for (int i = 0; i < TAGS; ++i) { v[i] = fmaxf(v[i], 0.0f); mx = fmaxf(mx, v[i]); }
    float sum = 0.0f;
#pragma unroll
    for (int i = 0; i < TAGS; ++i) { v[i] = __expf(v[i] - mx); sum += v[i]; }
    float inv = 1.0f / sum;
    float* orow = out + (size_t)t * TAGS;
#pragma unroll
    for (int i = 0; i < TAGS; ++i) orow[i] = v[i] * inv;
}

// ---------------------------------------------------------------------------
extern "C" void kernel_launch(void* const* d_in, const int* in_sizes, int n_in,
                              void* d_out, int out_size, void* d_ws, size_t ws_size,
                              hipStream_t stream) {
    const float* input_vecs = (const float*)d_in[0];
    const float* dp_table   = (const float*)d_in[1];
    const float* Wih0       = (const float*)d_in[2];
    const float* Whh0       = (const float*)d_in[3];
    const float* b0         = (const float*)d_in[4];
    const float* Wih1       = (const float*)d_in[5];
    const float* Whh1       = (const float*)d_in[6];
    const float* b1         = (const float*)d_in[7];
    const float* W_out      = (const float*)d_in[8];
    const float* b_out      = (const float*)d_in[9];
    const int*   dp_in      = (const int*)d_in[10];
    const int*   mask       = (const int*)d_in[11];
    float*       out        = (float*)d_out;

    // Workspace layout (all sizes 256B-multiples)
    char* ws = (char*)d_ws;
    __bf16* xb   = (__bf16*)(ws + 0);                       // 32768*800*2  = 52,428,800
    __bf16* Wb0  = (__bf16*)(ws + 52428800);                // 256*800*2    =    409,600
    float*  G0   = (float*)(ws + 52838400);                 // 32768*256*4  = 33,554,432
    __bf16* h0b  = (__bf16*)(ws + 86392832);                // 32768*64*2   =  4,194,304
    __bf16* Wb1  = (__bf16*)(ws + 90587136);                // 256*64*2     =     32,768
    float*  G1   = (float*)(ws + 90619904);                 // 32768*256*4  = 33,554,432
    __bf16* h1b  = (__bf16*)(ws + 124174336);               // 32768*64*2   =  4,194,304
    __bf16* Wob  = (__bf16*)(ws + 128368640);               // 64*64*2      =      8,192
    float*  tag  = (float*)(ws + 128376832);                // 32768*64*4   =  8,388,608

    // --- Stage 0: bf16 conversion / packing
    {
        size_t n = (size_t)T_LEN * XPAD;
        embed_kernel<<<(unsigned)((n + 255) / 256), 256, 0, stream>>>(input_vecs, dp_table, dp_in, xb);
    }
    cvt_pad_kernel<<<(256 * XPAD + 255) / 256, 256, 0, stream>>>(Wih0, Wb0, 256, 256, 774, XPAD);
    cvt_pad_kernel<<<(256 * 64 + 255) / 256, 256, 0, stream>>>(Wih1, Wb1, 256, 256, 64, 64);
    cvt_pad_kernel<<<(64 * 64 + 255) / 256, 256, 0, stream>>>(W_out, Wob, 64, TAGS, 64, 64);

    const int Mtiles = T_LEN / 16;   // 2048 (16-row tiles)

    // --- Layer 0: G0 = x @ Wih0^T + b0   (N=256 -> Nblk=4, K=800)
    {
        int waves = Mtiles * (NGATE / 64);           // 8192 waves, 8 per block
        gemm_bf16_wmma<<<waves / 8, 256, 0, stream>>>(xb, Wb0, b0, G0,
                                                      Mtiles, NGATE / 64, XPAD / 32,
                                                      XPAD, XPAD, NGATE, NGATE);
    }
    lstm_scan_kernel<<<2, 32, 0, stream>>>(G0, Whh0, h0b, T_LEN);

    // --- Layer 1: G1 = h0 @ Wih1^T + b1   (N=256 -> Nblk=4, K=64)
    {
        int waves = Mtiles * (NGATE / 64);
        gemm_bf16_wmma<<<waves / 8, 256, 0, stream>>>(h0b, Wb1, b1, G1,
                                                      Mtiles, NGATE / 64, 64 / 32,
                                                      64, 64, NGATE, NGATE);
    }
    lstm_scan_kernel<<<2, 32, 0, stream>>>(G1, Whh1, h1b, T_LEN);

    // --- Output projection: tag = h1 @ W_out^T + b_out  (N padded 64 -> Nblk=1)
    {
        int waves = Mtiles * (TAGPAD / 64);          // 2048 waves
        gemm_bf16_wmma<<<waves / 8, 256, 0, stream>>>(h1b, Wob, b_out, tag,
                                                      Mtiles, TAGPAD / 64, 64 / 32,
                                                      64, 64, TAGPAD, TAGS);
    }

    // --- mask / relu / softmax
    tag_softmax_kernel<<<(T_LEN + 255) / 256, 256, 0, stream>>>(tag, mask, out);
}
